// MoERouter_37374805410166
// MI455X (gfx1250) — matrix operand order repaired and checked by reading, not compiled, not run on hardware
//
#include <hip/hip_runtime.h>

// MoE router: logits = x @ W^T, softmax over E=64, top-2.
// x: [16384, 2048] f32   W: [64, 2048] f32
// out = concat(top_k_probs [16384,2], top_k_indices [16384,2] (as float),
//              probs [16384,64])

#define DDIM    2048
#define EEXP    64
#define NTOK    (4 * 4096)
#define NWAVES  8           // waves per block (256 threads)
#define WSTRIDE 2056        // bf16 row stride: dword stride 1028 == 4 mod 64 (bank spread)
#define LSTRIDE 65          // logits row stride in floats (65 mod 64 == 1)

typedef __attribute__((ext_vector_type(16))) __bf16   v16bf;
typedef __attribute__((ext_vector_type(2)))  __bf16   v2bf;
typedef __attribute__((ext_vector_type(8)))  float    v8f;
typedef __attribute__((ext_vector_type(4)))  float    v4f;
typedef __attribute__((ext_vector_type(4)))  unsigned v4u;

union BV { v16bf bf; v4u q[2]; };

__global__ __launch_bounds__(256, 1)
void moe_router_wmma_kernel(const float* __restrict__ x,
                            const float* __restrict__ w,
                            float* __restrict__ topp,
                            float* __restrict__ topi,
                            float* __restrict__ probs)
{
    // 257 KB W tile (bf16) + 32.5 KB logits + 6 KB partials: fits 320 KB WGP LDS.
    __shared__ __bf16 wlds[EEXP * WSTRIDE];
    __shared__ float  lg[NWAVES * 16 * LSTRIDE];
    __shared__ float  part[NWAVES * 16 * 2 * 6];

    const int tid = threadIdx.x;

    // ---- Stage gate weights fp32 -> bf16 into LDS (once per block) ----
    {
        const v4f* wv = (const v4f*)w;
        for (int i = tid; i < (EEXP * DDIM) / 4; i += 256) {
            v4f f = wv[i];
            int idx = i << 2;
            int row = idx >> 11;          // / DDIM
            int col = idx & (DDIM - 1);
            v2bf p0 = { (__bf16)f.x, (__bf16)f.y };
            v2bf p1 = { (__bf16)f.z, (__bf16)f.w };
            unsigned* dst = (unsigned*)&wlds[row * WSTRIDE + col];
            dst[0] = __builtin_bit_cast(unsigned, p0);
            dst[1] = __builtin_bit_cast(unsigned, p1);
        }
    }
    __syncthreads();

    const int wave   = tid >> 5;
    const int lane   = tid & 31;
    const int m      = lane & 15;   // row (A) / column-N (B,C) within tile
    const int hi     = lane >> 4;   // half-wave group
    const int token0 = (blockIdx.x * NWAVES + wave) * 16;

    const float* xrow = x + (size_t)(token0 + m) * DDIM;

    v8f acc[4] = {};   // 4 N-tiles of 16x16 f32 accumulators

    for (int kk = 0; kk < DDIM; kk += 32) {
        // ---- A: 16x32 bf16 tile, ISA lane layout ----
        // lanes 0-15 row m hold K = {0..7,16..23}+kk ; lanes 16-31 hold +8.
        const float* xp = xrow + kk + 8 * hi;
        v4f f0 = __builtin_nontemporal_load((const v4f*)(xp));        // K+0..3
        v4f f1 = __builtin_nontemporal_load((const v4f*)(xp + 4));    // K+4..7
        v4f f2 = __builtin_nontemporal_load((const v4f*)(xp + 16));   // K+16..19
        v4f f3 = __builtin_nontemporal_load((const v4f*)(xp + 20));   // K+20..23
        v16bf a;
        a[0]  = (__bf16)f0.x;  a[1]  = (__bf16)f0.y;
        a[2]  = (__bf16)f0.z;  a[3]  = (__bf16)f0.w;
        a[4]  = (__bf16)f1.x;  a[5]  = (__bf16)f1.y;
        a[6]  = (__bf16)f1.z;  a[7]  = (__bf16)f1.w;
        a[8]  = (__bf16)f2.x;  a[9]  = (__bf16)f2.y;
        a[10] = (__bf16)f2.z;  a[11] = (__bf16)f2.w;
        a[12] = (__bf16)f3.x;  a[13] = (__bf16)f3.y;
        a[14] = (__bf16)f3.z;  a[15] = (__bf16)f3.w;

        // ---- B: 32x16 bf16 tiles from LDS (expert = nt*16 + m) ----
        // lanes 0-15 hold K = {0..7,16..23}+kk ; lanes 16-31 hold +8:
        // two 8-bf16 runs per lane -> two ds_load_b128.
        const __bf16* wbase = wlds + (size_t)m * WSTRIDE + kk + 8 * hi;
        #pragma unroll
        for (int nt = 0; nt < 4; ++nt) {
            BV b;
            const __bf16* wp = wbase + nt * 16 * WSTRIDE;
            b.q[0] = *(const v4u*)(wp);
            b.q[1] = *(const v4u*)(wp + 16);
            acc[nt] = __builtin_amdgcn_wmma_f32_16x16x32_bf16(
                false, a, false, b.bf, (short)0, acc[nt], false, false);
        }
    }

    // ---- Scatter C tiles to LDS logits (token-major rows) ----
    // C layout: lane -> N=m, VGPR r -> token row r + 8*hi.
    float* lbase = lg + (size_t)(wave * 16) * LSTRIDE;
    #pragma unroll
    for (int nt = 0; nt < 4; ++nt) {
        #pragma unroll
        for (int r = 0; r < 8; ++r) {
            lbase[(r + 8 * hi) * LSTRIDE + nt * 16 + m] = acc[nt][r];
        }
    }
    __syncthreads();

    // ---- Softmax + top-2: 2 lanes per token, 32 experts each ----
    const float* lr = lg + (size_t)(wave * 16 + m) * LSTRIDE;
    const int e0 = 32 * hi;
    {
        // fused local max + local top-2 (top-1 IS the local max)
        float l0 = -3.4e38f, l1 = -3.4e38f;
        int   i0 = e0,       i1 = e0;
        for (int e = 0; e < 32; ++e) {
            float v = lr[e0 + e];
            if (v > l0)      { l1 = l0; i1 = i0; l0 = v; i0 = e0 + e; }
            else if (v > l1) { l1 = v;  i1 = e0 + e; }
        }
        float sum = 0.0f;
        for (int e = 0; e < 32; ++e) sum += __expf(lr[e0 + e] - l0);

        float* pp = part + (size_t)(((wave * 16 + m) * 2) + hi) * 6;
        pp[0] = l0;  pp[1] = sum;
        pp[2] = l0;  pp[3] = l1;
        pp[4] = (float)i0;  pp[5] = (float)i1;
    }
    __syncthreads();
    {
        const float* ps = part + (size_t)(((wave * 16 + m) * 2) + hi) * 6;
        const float* po = part + (size_t)(((wave * 16 + m) * 2) + (1 - hi)) * 6;
        float mxs = ps[0], sums = ps[1];
        float mxo = po[0], sumo = po[1];
        float M   = fmaxf(mxs, mxo);
        float S   = sums * __expf(mxs - M) + sumo * __expf(mxo - M);
        float inv = 1.0f / S;

        const int token = token0 + m;
        float* pr = probs + (size_t)token * EEXP + e0;
        for (int e = 0; e < 32; ++e)
            __builtin_nontemporal_store(__expf(lr[e0 + e] - M) * inv, pr + e);

        if (hi == 0) {
            // A = experts 0..31 (self), B = experts 32..63 (other half)
            float a0 = ps[2], a1 = ps[3];
            int  ia0 = (int)ps[4], ia1 = (int)ps[5];
            float b0 = po[2], b1 = po[3];
            int  ib0 = (int)po[4], ib1 = (int)po[5];
            float L0, L1; int I0, I1;
            if (a0 >= b0) {                      // tie -> lower index (A side)
                L0 = a0; I0 = ia0;
                if (a1 >= b0) { L1 = a1; I1 = ia1; }
                else          { L1 = b0; I1 = ib0; }
            } else {
                L0 = b0; I0 = ib0;
                if (a0 >= b1) { L1 = a0; I1 = ia0; }
                else          { L1 = b1; I1 = ib1; }
            }
            float q0 = __expf(L0 - M), q1 = __expf(L1 - M);
            float s2inv = 1.0f / (q0 + q1);
            topp[token * 2 + 0] = q0 * s2inv;
            topp[token * 2 + 1] = q1 * s2inv;
            topi[token * 2 + 0] = (float)I0;
            topi[token * 2 + 1] = (float)I1;
        }
    }
}

extern "C" void kernel_launch(void* const* d_in, const int* in_sizes, int n_in,
                              void* d_out, int out_size, void* d_ws, size_t ws_size,
                              hipStream_t stream) {
    (void)in_sizes; (void)n_in; (void)out_size; (void)d_ws; (void)ws_size;
    const float* x = (const float*)d_in[0];
    const float* w = (const float*)d_in[1];
    // d_in[2] is top_k (==2), compile-time constant here.
    float* out   = (float*)d_out;
    float* topp  = out;               // [NTOK, 2]
    float* topi  = out + 2 * NTOK;    // [NTOK, 2] (indices as float values)
    float* probs = out + 4 * NTOK;    // [NTOK, 64]

    dim3 grid(NTOK / (16 * NWAVES)); // 128 blocks
    dim3 block(256);                 // 8 wave32
    moe_router_wmma_kernel<<<grid, block, 0, stream>>>(x, w, topp, topi, probs);
}
